// CausalGraphLearner_67095979099074
// MI455X (gfx1250) — compile-verified, matrix operand order checked by reading, not run on hardware
//
#include <hip/hip_runtime.h>

// ---------------- types ----------------
typedef __attribute__((ext_vector_type(16))) __bf16 v16bf;
typedef __attribute__((ext_vector_type(8)))  float  v8f;

#define B_  8
#define N_  256
#define D_  64
#define H_  256
#define K2_ 128           // H/2
#define KAUG_ 128         // augmented K for GEMM1: [x*y prod | y]
#define LN_EPS 1e-5f
#define INV_SQRT2 0.70710678118654752f

// native f32 -> bf16 (RNE); backend pairs these into v_cvt_pk_bf16_f32
static __device__ __forceinline__ unsigned short f2bf(float f) {
    __bf16 h = (__bf16)f;
    return __builtin_bit_cast(unsigned short, h);
}

union TileU {
    unsigned short u[16];
    unsigned int   w[8];
    uint4          q[2];
    v16bf          v;
};

// Load a 16-bit WMMA operand tile (A or B role) from a row-major [row][k]
// array of bf16. CDNA5 16-bit operand layout: lane L owns row (rowBase+L%16);
// lane-group (L/16) selects K half; the 16 elements are two contiguous
// 8-element chunks at kBase+8*half and kBase+8*half+16.
static __device__ __forceinline__ v16bf load_bf16_tile(const unsigned short* base,
                                                       int row, int rowStride,
                                                       int kBase, int halfsel) {
    const unsigned short* p = base + row * rowStride + kBase + halfsel * 8;
    TileU t;
    t.q[0] = *reinterpret_cast<const uint4*>(p);
    t.q[1] = *reinterpret_cast<const uint4*>(p + 16);
    return t.v;
}

static __device__ __forceinline__ float gelu_exact(float v) {
    return 0.5f * v * (1.0f + erff(v * INV_SQRT2));
}

// ---------------- prep kernels ----------------
// rowA[bx][h] = b1[h] + sum_d slots[bx][d] * (Wa[d][h] + Wc[d][h])
__global__ __launch_bounds__(256) void prep_rowA(const float* __restrict__ slots,
                                                 const float* __restrict__ W1,
                                                 const float* __restrict__ b1,
                                                 float* __restrict__ rowA) {
    int idx = blockIdx.x * 256 + threadIdx.x;   // B*N*H = 524288 threads
    int h  = idx & (H_ - 1);
    int bx = idx >> 8;
    const float* sr = slots + bx * D_;
    float acc = b1[h];
#pragma unroll 8
    for (int d = 0; d < D_; ++d)
        acc += sr[d] * (W1[d * H_ + h] + W1[(128 + d) * H_ + h]);
    rowA[idx] = acc;
}

// WaugT[h][k] (256x128 bf16):  k<64 -> Wd[k][h] ; k>=64 -> Wb[k-64][h]-Wc[k-64][h]
// W2T[n][k]  (128x256 bf16):   W2[k][n]
__global__ __launch_bounds__(256) void prep_weights(const float* __restrict__ W1,
                                                    const float* __restrict__ W2,
                                                    unsigned short* __restrict__ waugT,
                                                    unsigned short* __restrict__ w2T) {
    int tid = blockIdx.x * 256 + threadIdx.x;   // 65536 threads
    if (tid < H_ * KAUG_) {
        int h = tid >> 7, k = tid & 127;
        float v;
        if (k < 64) v = W1[(192 + k) * H_ + h];
        else        v = W1[k * H_ + h] - W1[(k + 64) * H_ + h];
        waugT[tid] = f2bf(v);
    } else {
        int t2 = tid - H_ * KAUG_;
        int n = t2 >> 8, k = t2 & 255;
        w2T[t2] = f2bf(W2[k * K2_ + n]);
    }
}

// ---------------- main kernel: one WG per (b,x) ----------------
__global__ __launch_bounds__(256) void cgl_main(const float* __restrict__ slots,
                                                const float* __restrict__ ln_g,
                                                const float* __restrict__ ln_b,
                                                const float* __restrict__ b2,
                                                const float* __restrict__ W3,
                                                const float* __restrict__ b3,
                                                const float* __restrict__ rowA,
                                                const unsigned short* __restrict__ waugT,
                                                const unsigned short* __restrict__ w2T,
                                                float* __restrict__ out) {
    __shared__ __align__(16) float          hbuf[32 * H_];       // 32KB: GEMM1 out; reused: out2 + red
    __shared__ __align__(16) unsigned short gbuf[32 * H_];       // 16KB: LN+gelu result (bf16) = GEMM2 A
    __shared__ __align__(16) unsigned short augL[32 * KAUG_];    //  8KB: staged GEMM1 A chunk (bf16)
    __shared__ float sxL[D_];
    __shared__ float rowAL[H_];
    __shared__ float lngL[H_], lnbL[H_];
    __shared__ float b2L[K2_], w3L[K2_];

    const int t    = threadIdx.x;
    const int lane = t & 31;
    const int wave = t >> 5;
    const int halfsel = lane >> 4;
    const int l16  = lane & 15;

    const int bx = blockIdx.x;              // b*256 + x
    const int b  = bx >> 8;
    const int x  = bx & 255;
    const float* slotsB = slots + (size_t)b * N_ * D_;

    if (t < D_) sxL[t] = slotsB[x * D_ + t];
    rowAL[t] = rowA[(size_t)bx * H_ + t];
    lngL[t]  = ln_g[t];
    lnbL[t]  = ln_b[t];
    if (t < K2_) { b2L[t] = b2[t]; w3L[t] = W3[t]; }
    const float b3v = b3[0];
    __syncthreads();

    const int mt  = wave & 1;               // M sub-tile (16 rows) within 32-row chunk
    const int ng  = (wave >> 1) * 64;       // GEMM1 N strip base (64 cols)
    const int ng2 = (wave >> 1) * 32;       // GEMM2 N strip base (32 cols)

    // staging coords: thread t owns row (t/8), 16-wide k-chunk ((t%8)*16)
    const int sRow = t >> 3;
    const int sKb  = (t & 7) * 16;

    for (int c = 0; c < 8; ++c) {
        // ---- stage augmented A chunk (32x128 bf16) into LDS, once per chunk ----
        {
            const float* sr = slotsB + (c * 32 + sRow) * D_;
            TileU tu;
            if (sKb < 64) {
#pragma unroll
                for (int e = 0; e < 16; ++e) tu.u[e] = f2bf(sr[sKb + e] * sxL[sKb + e]);
            } else {
#pragma unroll
                for (int e = 0; e < 16; ++e) tu.u[e] = f2bf(sr[sKb - 64 + e]);
            }
            uint4* dst = reinterpret_cast<uint4*>(augL + t * 16);
            dst[0] = tu.q[0];
            dst[1] = tu.q[1];
        }
        if (c < 7) __builtin_prefetch(slotsB + (c + 1) * 32 * D_ + t * 8, 0, 1);
        __syncthreads();

        // ---- GEMM1: Aug(32x128) @ Waug(128x256), bf16 WMMA, f32 acc ----
        v8f acc[4];
#pragma unroll
        for (int nt = 0; nt < 4; ++nt) acc[nt] = (v8f){0,0,0,0,0,0,0,0};
#pragma unroll
        for (int kk = 0; kk < 4; ++kk) {
            const int kBase = kk * 32;
            v16bf aT = load_bf16_tile(augL, mt * 16 + l16, KAUG_, kBase, halfsel);
#pragma unroll
            for (int nt = 0; nt < 4; ++nt) {
                v16bf bT = load_bf16_tile(waugT, ng + nt * 16 + l16, KAUG_, kBase, halfsel);
                acc[nt] = __builtin_amdgcn_wmma_f32_16x16x32_bf16(
                    false, aT, false, bT, (short)0, acc[nt], false, false);
            }
        }
        // epilogue: add rowA (broadcast over y), spill f32 to LDS
        const int mBaseL = mt * 16 + (halfsel ? 8 : 0);
#pragma unroll
        for (int nt = 0; nt < 4; ++nt) {
            const int col = ng + nt * 16 + l16;
            const float add = rowAL[col];
#pragma unroll
            for (int r = 0; r < 8; ++r)
                hbuf[(mBaseL + r) * H_ + col] = acc[nt][r] + add;
        }
        __syncthreads();

        // ---- LayerNorm over H + exact GELU, emit bf16 ----
#pragma unroll
        for (int it = 0; it < 4; ++it) {
            const int row = it * 8 + wave;
            float v[8];
            float s = 0.0f;
#pragma unroll
            for (int j = 0; j < 8; ++j) { v[j] = hbuf[row * H_ + lane + j * 32]; s += v[j]; }
#pragma unroll
            for (int off = 16; off >= 1; off >>= 1) s += __shfl_xor(s, off);
            const float mu = s * (1.0f / 256.0f);
            float q = 0.0f;
#pragma unroll
            for (int j = 0; j < 8; ++j) { float d = v[j] - mu; q += d * d; }
#pragma unroll
            for (int off = 16; off >= 1; off >>= 1) q += __shfl_xor(q, off);
            const float rs = rsqrtf(q * (1.0f / 256.0f) + LN_EPS);
#pragma unroll
            for (int j = 0; j < 8; ++j) {
                const int hc = lane + j * 32;
                float g = (v[j] - mu) * rs * lngL[hc] + lnbL[hc];
                gbuf[row * H_ + hc] = f2bf(gelu_exact(g));
            }
        }
        __syncthreads();

        // ---- GEMM2: g(32x256) @ W2(256x128), bf16 WMMA, f32 acc ----
        v8f acc2[2];
        acc2[0] = (v8f){0,0,0,0,0,0,0,0};
        acc2[1] = (v8f){0,0,0,0,0,0,0,0};
        const unsigned short* garow = gbuf + (mt * 16 + l16) * H_;
#pragma unroll
        for (int kk = 0; kk < 8; ++kk) {
            const int kBase = kk * 32;
            v16bf aT = load_bf16_tile(garow, 0, 0, kBase, halfsel);
#pragma unroll
            for (int tt = 0; tt < 2; ++tt) {
                v16bf bT = load_bf16_tile(w2T, ng2 + tt * 16 + l16, H_, kBase, halfsel);
                acc2[tt] = __builtin_amdgcn_wmma_f32_16x16x32_bf16(
                    false, aT, false, bT, (short)0, acc2[tt], false, false);
            }
        }
        // epilogue: +b2, GELU, store f32 into hbuf reused as out2 (32x128)
        float* out2 = hbuf;
#pragma unroll
        for (int tt = 0; tt < 2; ++tt) {
            const int col = ng2 + tt * 16 + l16;
            const float badd = b2L[col];
#pragma unroll
            for (int r = 0; r < 8; ++r) {
                float vv = acc2[tt][r] + badd;
                out2[(mt * 16 + (halfsel ? 8 : 0) + r) * K2_ + col] = gelu_exact(vv);
            }
        }
        __syncthreads();

        // ---- final: dot with W3, +b3, sigmoid ----
        float* red = hbuf + 32 * K2_;        // 256 floats of scratch past out2
        {
            const int row = t >> 3, seg = t & 7;
            float p = 0.0f;
#pragma unroll
            for (int k = 0; k < 16; ++k)
                p += out2[row * K2_ + seg * 16 + k] * w3L[seg * 16 + k];
            red[row * 8 + seg] = p;
        }
        __syncthreads();
        if (t < 32) {
            float s = b3v;
#pragma unroll
            for (int sg = 0; sg < 8; ++sg) s += red[t * 8 + sg];
            out[(size_t)bx * N_ + c * 32 + t] = 1.0f / (1.0f + expf(-s));
        }
        __syncthreads();
    }
}

// ---------------- launcher ----------------
extern "C" void kernel_launch(void* const* d_in, const int* in_sizes, int n_in,
                              void* d_out, int out_size, void* d_ws, size_t ws_size,
                              hipStream_t stream) {
    const float* slots = (const float*)d_in[0];
    const float* W1    = (const float*)d_in[1];
    const float* b1    = (const float*)d_in[2];
    const float* ln_g  = (const float*)d_in[3];
    const float* ln_b  = (const float*)d_in[4];
    const float* W2    = (const float*)d_in[5];
    const float* b2    = (const float*)d_in[6];
    const float* W3    = (const float*)d_in[7];
    const float* b3    = (const float*)d_in[8];
    float* out = (float*)d_out;

    char* ws = (char*)d_ws;
    float*          rowA  = (float*)ws;                                   // B*N*H f32 = 2MB
    unsigned short* waugT = (unsigned short*)(ws + (size_t)B_*N_*H_*4);   // 256x128 bf16
    unsigned short* w2T   = (unsigned short*)(ws + (size_t)B_*N_*H_*4 + (size_t)H_*KAUG_*2);

    prep_rowA   <<<B_*N_*H_/256, 256, 0, stream>>>(slots, W1, b1, rowA);
    prep_weights<<<(H_*KAUG_ + K2_*H_)/256, 256, 0, stream>>>(W1, W2, waugT, w2T);
    cgl_main    <<<B_*N_, 256, 0, stream>>>(slots, ln_g, ln_b, b2, W3, b3,
                                            rowA, waugT, w2T, out);
}
